// ModuleGAT_9122510537162
// MI455X (gfx1250) — matrix operand order repaired and checked by reading, not compiled, not run on hardware
//
#include <hip/hip_runtime.h>
#include <hip/hip_bf16.h>

typedef __attribute__((ext_vector_type(16))) _Float16 v16h;
typedef __attribute__((ext_vector_type(8)))  _Float16 v8h;
typedef __attribute__((ext_vector_type(8)))  float    v8f;
typedef __attribute__((ext_vector_type(4)))  unsigned v4u;
typedef __attribute__((ext_vector_type(8)))  int      v8i;
typedef __attribute__((ext_vector_type(4)))  int      v4i;

#define N_NODES 50000
#define N_EDGES 800000
#define F_DIM   128
#define HEADS   4
#define OUT_C   32
#define E2      (N_EDGES + N_NODES)

// ---------- ordered float <-> uint encoding for atomicMax on floats ----------
__device__ __forceinline__ unsigned enc_f32(float f) {
    unsigned u = __float_as_uint(f);
    return (u & 0x80000000u) ? ~u : (u ^ 0x80000000u);
}
__device__ __forceinline__ float dec_f32(unsigned e) {
    unsigned u = (e & 0x80000000u) ? (e ^ 0x80000000u) : ~e;
    return __uint_as_float(u);
}

// ---------- utility kernels ----------
__global__ void zero_u32(unsigned* __restrict__ p, int n) {
    int i = blockIdx.x * blockDim.x + threadIdx.x;
    if (i < n) p[i] = 0u;
}

__global__ void cvt_f16(const float* __restrict__ x, _Float16* __restrict__ y, int n) {
    int i = blockIdx.x * blockDim.x + threadIdx.x;
    if (i < n) y[i] = (_Float16)x[i];
}

// Pack W[128][128] (f32 row-major, K x Ncol) into WMMA B-fragment order:
// Wb[((kt*8+nt)*32+lane)*16 + j] = W[kt*32 + (lane>=16?16:0) + j][nt*16 + (lane&15)]
__global__ void pack_w(const float* __restrict__ W, _Float16* __restrict__ Wb) {
    int tid = blockIdx.x * blockDim.x + threadIdx.x;   // 16384 total
    if (tid >= F_DIM * F_DIM) return;
    int j    = tid & 15;
    int lane = (tid >> 4) & 31;
    int nt   = (tid >> 9) & 7;
    int kt   = tid >> 12;
    int col  = nt * 16 + (lane & 15);
    int k    = kt * 32 + ((lane >= 16) ? 16 : 0) + j;
    Wb[tid] = (_Float16)W[k * F_DIM + col];
}

// ---------- TDM: DMA the 32KB packed-B buffer into LDS (one descriptor) ----------
// D# per CDNA5 ISA ch.8: group0 = {count, lds_addr, global_addr, type=2};
// group1 = {data_size=8B, tensor_dim0=tile_dim0=4096, tensor_dim1=tile_dim1=1,
//           tensor_dim0_stride=4096}; groups 2/3 zero (<=2D tensor).
__device__ __forceinline__ void tdm_load_wb(const _Float16* Wb, unsigned lds_byte_off) {
    unsigned long long ga = (unsigned long long)(const void*)Wb;
    v4u g0;
    g0[0] = 1u;                                               // count=1, no gather
    g0[1] = lds_byte_off;                                     // LDS dest byte address
    g0[2] = (unsigned)(ga & 0xFFFFFFFFu);                     // global_addr[31:0]
    g0[3] = (unsigned)((ga >> 32) & 0x1FFFFFFu) | (2u << 30); // addr[56:32] | type=2
    v8i g1;
    g1[0] = (int)(3u << 16);                                  // data_size=8B, mask=0
    g1[1] = (int)(4096u << 16);                               // tensor_dim0[15:0]=4096
    g1[2] = (int)(1u << 16);                                  // tensor_dim1[15:0]=1
    g1[3] = (int)(4096u << 16);                               // tile_dim0=4096
    g1[4] = 1;                                                // tile_dim1=1, tile_dim2=0
    g1[5] = 4096;                                             // tensor_dim0_stride[31:0]
    g1[6] = 0;
    g1[7] = 0;
    v4i gz = {0, 0, 0, 0};
#if defined(__clang_major__) && (__clang_major__ >= 23)
    v8i gz8 = {0, 0, 0, 0, 0, 0, 0, 0};
    __builtin_amdgcn_tensor_load_to_lds(g0, g1, gz, gz, gz8, 0);
#else
    __builtin_amdgcn_tensor_load_to_lds(g0, g1, gz, gz, 0);
#endif
    __builtin_amdgcn_s_wait_tensorcnt(0);
}

// ---------- WMMA GEMM: C[nrows,128] = A[nrows,128](f16) * W[128,128](f16 packed) ----------
// 256 threads = 8 waves; each wave computes a 16-row strip x 128 cols (8 tiles).
__global__ __launch_bounds__(256) void gemm_wmma(
        const _Float16* __restrict__ A, const _Float16* __restrict__ Wb,
        float* __restrict__ C, int nrows) {
    __shared__ _Float16 lds[F_DIM * F_DIM];   // 32 KB packed B fragments

    if ((threadIdx.x >> 5) == 0) {            // wave 0 issues one TDM DMA, drains it
        // low 32 bits of the generic address of an LDS object = LDS byte offset
        unsigned ldsOff = (unsigned)(unsigned long long)(void*)&lds[0];
        tdm_load_wb(Wb, ldsOff);
    }
    __syncthreads();
    // Make the TDM's write to LDS visible to the optimizer: the pointer escapes
    // into opaque asm with a memory clobber, so the ds_loads below cannot be
    // folded away as reads of never-written memory.
    asm volatile("" :: "v"((void*)&lds[0]) : "memory");

    const int wave    = threadIdx.x >> 5;
    const int lane    = threadIdx.x & 31;
    const int rowBase = blockIdx.x * 128 + wave * 16;
    const int hiHalf  = lane >> 4;                 // 0 or 1
    int lrow = rowBase + (lane & 15);
    if (lrow > nrows - 1) lrow = nrows - 1;        // clamp OOB loads (stores guarded)

    v8f acc[8] = {};

    #pragma unroll
    for (int kt = 0; kt < 4; ++kt) {
        // Batch-load all 8 B fragments of this K-chunk so the ds_load clause
        // overlaps the previous chunk's WMMAs instead of stalling each one.
        v16h bfr[8];
        #pragma unroll
        for (int nt = 0; nt < 8; ++nt)
            bfr[nt] = *(const v16h*)(&lds[((kt * 8 + nt) * 32 + lane) * 16]);

        // A fragment: lane<16 holds K {0..15}; lane>=16 holds K {16..31}
        const _Float16* ap = A + (size_t)lrow * F_DIM + kt * 32 + hiHalf * 16;
        v8h alo = *(const v8h*)(ap);
        v8h ahi = *(const v8h*)(ap + 8);
        v16h a  = __builtin_shufflevector(alo, ahi,
                    0, 1, 2, 3, 4, 5, 6, 7, 8, 9, 10, 11, 12, 13, 14, 15);

        #pragma unroll
        for (int nt = 0; nt < 8; ++nt)
            acc[nt] = __builtin_amdgcn_wmma_f32_16x16x32_f16(
                          false, a, false, bfr[nt], (short)0, acc[nt], false, false);
    }

    // C/D layout: VGPR r, lanes 0-15 -> M=r, N=lane; lanes 16-31 -> M=8+r, N=lane-16
    const int rOff = hiHalf * 8;
    if (rowBase + 16 <= nrows) {
        // fast path: whole 16-row strip in range (all but the last block)
        #pragma unroll
        for (int nt = 0; nt < 8; ++nt) {
            int col = nt * 16 + (lane & 15);
            #pragma unroll
            for (int r = 0; r < 8; ++r)
                C[(size_t)(rowBase + rOff + r) * F_DIM + col] = acc[nt][r];
        }
    } else {
        #pragma unroll
        for (int nt = 0; nt < 8; ++nt) {
            int col = nt * 16 + (lane & 15);
            #pragma unroll
            for (int r = 0; r < 8; ++r) {
                int row = rowBase + rOff + r;
                if (row < nrows) C[(size_t)row * F_DIM + col] = acc[nt][r];
            }
        }
    }
}

// ---------- attention coefficients: a_src[n,h] = sum_c h[n,h,c]*att_src[h,c] ----------
__global__ void attn_coef(const float* __restrict__ h,
                          const float* __restrict__ att_src,
                          const float* __restrict__ att_dst,
                          float* __restrict__ as, float* __restrict__ ad) {
    int tid = blockIdx.x * blockDim.x + threadIdx.x;
    if (tid >= N_NODES * HEADS) return;
    int n = tid >> 2, hd = tid & 3;
    const float* hp = h + (size_t)n * F_DIM + hd * OUT_C;
    const float* ps = att_src + hd * OUT_C;
    const float* pd = att_dst + hd * OUT_C;
    float s = 0.f, d = 0.f;
    #pragma unroll
    for (int c = 0; c < OUT_C; ++c) { float v = hp[c]; s += v * ps[c]; d += v * pd[c]; }
    as[tid] = s;
    ad[tid] = d;
}

__device__ __forceinline__ void edge_src_dst(const long long* __restrict__ ei,
                                             int e, int& src, int& dst) {
    if (e < N_EDGES) { src = (int)ei[e]; dst = (int)ei[N_EDGES + e]; }
    else             { src = dst = e - N_EDGES; }
}

// ---------- edge pass 1: e = leakyrelu(a_src[src]+a_dst[dst], 0.2); segment max ----------
__global__ void edge_max(const long long* __restrict__ ei,
                         const float* __restrict__ as, const float* __restrict__ ad,
                         float* __restrict__ ebuf, unsigned* __restrict__ menc) {
    int e = blockIdx.x * blockDim.x + threadIdx.x;
    if (e >= E2) return;
    int src, dst; edge_src_dst(ei, e, src, dst);
    #pragma unroll
    for (int hd = 0; hd < HEADS; ++hd) {
        float v = as[src * HEADS + hd] + ad[dst * HEADS + hd];
        v = v > 0.f ? v : 0.2f * v;
        ebuf[(size_t)e * HEADS + hd] = v;
        atomicMax(&menc[dst * HEADS + hd], enc_f32(v));
    }
}

// ---------- edge pass 2: ex = exp(e - m[dst]); segment sum ----------
__global__ void edge_exp(const long long* __restrict__ ei,
                         const unsigned* __restrict__ menc,
                         float* __restrict__ ebuf, float* __restrict__ ssum) {
    int e = blockIdx.x * blockDim.x + threadIdx.x;
    if (e >= E2) return;
    int src, dst; edge_src_dst(ei, e, src, dst);
    (void)src;
    #pragma unroll
    for (int hd = 0; hd < HEADS; ++hd) {
        float ex = __expf(ebuf[(size_t)e * HEADS + hd] - dec_f32(menc[dst * HEADS + hd]));
        ebuf[(size_t)e * HEADS + hd] = ex;
        atomicAdd(&ssum[dst * HEADS + hd], ex);
    }
}

// ---------- edge pass 3: agg[dst] += h[src] * alpha  (32 lanes/edge, 4 ch/lane) ----------
__global__ void edge_agg(const long long* __restrict__ ei,
                         const float* __restrict__ ebuf, const float* __restrict__ ssum,
                         const float* __restrict__ h, float* __restrict__ agg) {
    int tid = blockIdx.x * blockDim.x + threadIdx.x;
    int e = tid >> 5, lane = tid & 31;
    if (e >= E2) return;
    int src, dst; edge_src_dst(ei, e, src, dst);
    int hd = lane >> 3;                                    // channel group -> head
    float alpha = ebuf[(size_t)e * HEADS + hd] /
                  (ssum[dst * HEADS + hd] + 1e-16f);
    const float4 hv = *(const float4*)(h + (size_t)src * F_DIM + lane * 4);
    float* op = agg + (size_t)dst * F_DIM + lane * 4;
    atomicAdd(op + 0, hv.x * alpha);
    atomicAdd(op + 1, hv.y * alpha);
    atomicAdd(op + 2, hv.z * alpha);
    atomicAdd(op + 3, hv.w * alpha);
}

// ---------- bias + leakyrelu(0.1); optional f32 and f16 outputs ----------
__global__ void bias_act(const float* __restrict__ agg, const float* __restrict__ bias,
                         float* __restrict__ of32, _Float16* __restrict__ of16, int n) {
    int i = blockIdx.x * blockDim.x + threadIdx.x;
    if (i >= n) return;
    float v = agg[i] + bias[i & (F_DIM - 1)];
    v = v > 0.f ? v : 0.1f * v;
    if (of32) of32[i] = v;
    if (of16) of16[i] = (_Float16)v;
}

extern "C" void kernel_launch(void* const* d_in, const int* in_sizes, int n_in,
                              void* d_out, int out_size, void* d_ws, size_t ws_size,
                              hipStream_t stream) {
    const float*     x    = (const float*)d_in[0];
    const long long* ei   = (const long long*)d_in[1];
    const float*     W0   = (const float*)d_in[2];
    const float*     as0  = (const float*)d_in[3];
    const float*     ad0  = (const float*)d_in[4];
    const float*     b0   = (const float*)d_in[5];
    const float*     W1   = (const float*)d_in[6];
    const float*     as1  = (const float*)d_in[7];
    const float*     ad1  = (const float*)d_in[8];
    const float*     b1   = (const float*)d_in[9];

    // ---- carve workspace ----
    char* base = (char*)d_ws;
    size_t off = 0;
    auto carve = [&](size_t bytes) -> void* {
        void* p = base + off;
        off = (off + bytes + 255) & ~(size_t)255;
        return p;
    };
    _Float16* hf16 = (_Float16*)carve((size_t)N_NODES * F_DIM * 2);
    _Float16* Wb   = (_Float16*)carve((size_t)F_DIM * F_DIM * 2);
    float*    hbuf = (float*)   carve((size_t)N_NODES * F_DIM * 4);
    float*    agg  = (float*)   carve((size_t)N_NODES * F_DIM * 4);
    float*    acs  = (float*)   carve((size_t)N_NODES * HEADS * 4);
    float*    acd  = (float*)   carve((size_t)N_NODES * HEADS * 4);
    unsigned* menc = (unsigned*)carve((size_t)N_NODES * HEADS * 4);
    float*    ssum = (float*)   carve((size_t)N_NODES * HEADS * 4);
    float*    ebuf = (float*)   carve((size_t)E2 * HEADS * 4);

    const int TB = 256;
    auto blocks = [](long long n, int tb) { return (unsigned)((n + tb - 1) / tb); };

    auto layer = [&](const float* W, const float* att_s, const float* att_d,
                     const float* bias, float* outF32, _Float16* outF16) {
        pack_w<<<blocks(F_DIM * F_DIM, TB), TB, 0, stream>>>(W, Wb);
        gemm_wmma<<<blocks(N_NODES, 128), TB, 0, stream>>>(hf16, Wb, hbuf, N_NODES);
        attn_coef<<<blocks((long long)N_NODES * HEADS, TB), TB, 0, stream>>>(
            hbuf, att_s, att_d, acs, acd);
        zero_u32<<<blocks((long long)N_NODES * HEADS, TB), TB, 0, stream>>>(menc, N_NODES * HEADS);
        zero_u32<<<blocks((long long)N_NODES * HEADS, TB), TB, 0, stream>>>(
            (unsigned*)ssum, N_NODES * HEADS);
        zero_u32<<<blocks((long long)N_NODES * F_DIM, TB), TB, 0, stream>>>(
            (unsigned*)agg, N_NODES * F_DIM);
        edge_max<<<blocks(E2, TB), TB, 0, stream>>>(ei, acs, acd, ebuf, menc);
        edge_exp<<<blocks(E2, TB), TB, 0, stream>>>(ei, menc, ebuf, ssum);
        edge_agg<<<blocks((long long)E2 * 32, TB), TB, 0, stream>>>(ei, ebuf, ssum, hbuf, agg);
        bias_act<<<blocks((long long)N_NODES * F_DIM, TB), TB, 0, stream>>>(
            agg, bias, outF32, outF16, N_NODES * F_DIM);
    };

    // layer 0: input x -> f16; output (leakyrelu'd) written back into hf16 for layer 1
    cvt_f16<<<blocks((long long)N_NODES * F_DIM, TB), TB, 0, stream>>>(
        x, hf16, N_NODES * F_DIM);
    layer(W0, as0, ad0, b0, /*f32 out*/ nullptr, /*f16 out*/ hf16);
    // layer 1: final output f32 -> d_out
    layer(W1, as1, ad1, b1, (float*)d_out, nullptr);

    (void)in_sizes; (void)n_in; (void)out_size; (void)ws_size;
}
